// VQState_25881472925810
// MI455X (gfx1250) — matrix operand order repaired and checked by reading, not compiled, not run on hardware
//
#include <hip/hip_runtime.h>
#include <hip/hip_bf16.h>

#define NTOK 16384
#define DF   256
#define KC   4096
#define GAMMA_F 0.99f
#define MTOK 128              // tokens per block (8 waves x 16)
#define NT   (KC / 16)        // 256 code tiles

typedef __attribute__((ext_vector_type(16))) _Float16 v16h;
typedef __attribute__((ext_vector_type(8)))  _Float16 v8h;
typedef __attribute__((ext_vector_type(8)))  float    v8f;
typedef __attribute__((__vector_size__(16))) int      v4i;

#if __has_builtin(__builtin_amdgcn_global_load_async_to_lds_b128) && \
    __has_builtin(__builtin_amdgcn_s_wait_asynccnt)
#define USE_ASYNC_LDS 1
#else
#define USE_ASYNC_LDS 0
#endif

// Order-preserving key: high 32 = monotone-mapped float bits, low 32 = index.
// min(key) == (min distance, then min index)  -> matches jnp.argmin tie rule.
__device__ __forceinline__ unsigned long long packKey(float d, unsigned idx) {
    unsigned u = __float_as_uint(d);
    u ^= (u & 0x80000000u) ? 0xFFFFFFFFu : 0x80000000u;
    return (((unsigned long long)u) << 32) | (unsigned long long)idx;
}

// Stage one 8KB B tile (16 codes x 256 halfs, contiguous) into LDS.
// Async DMA path when the CDNA5 builtins exist; reg+ds_store fallback otherwise.
__device__ __forceinline__ void stage_tile(_Float16* dst, const _Float16* src, int tid)
{
#if USE_ASYNC_LDS
    const char* g = (const char*)src;
    char*       l = (char*)dst;
    __builtin_amdgcn_global_load_async_to_lds_b128(
        (__attribute__((address_space(1))) v4i*)(g + tid * 16),
        (__attribute__((address_space(3))) v4i*)(l + tid * 16), 0, 0);
    __builtin_amdgcn_global_load_async_to_lds_b128(
        (__attribute__((address_space(1))) v4i*)(g + 4096 + tid * 16),
        (__attribute__((address_space(3))) v4i*)(l + 4096 + tid * 16), 0, 0);
#else
    uint4*       d = (uint4*)dst;
    const uint4* s = (const uint4*)src;
    d[tid]       = s[tid];
    d[tid + 256] = s[tid + 256];
#endif
}

__device__ __forceinline__ void wait_stage()
{
#if USE_ASYNC_LDS
    __builtin_amdgcn_s_wait_asynccnt(0);
#endif
}

// ---- prep: f32 row -> f16 copy + squared-norm (+ optional mask count) ----
__global__ __launch_bounds__(256)
void vq_prep_rows(const float* __restrict__ src, _Float16* __restrict__ dsth,
                  float* __restrict__ sq, const int* __restrict__ mask,
                  int* __restrict__ totalCnt)
{
    __shared__ float sred[8];
    const int row = blockIdx.x;
    const int d   = threadIdx.x;
    float v = src[(size_t)row * DF + d];
    dsth[(size_t)row * DF + d] = (_Float16)v;
    float s = v * v;
#pragma unroll
    for (int off = 16; off > 0; off >>= 1) s += __shfl_xor(s, off, 32);
    if ((threadIdx.x & 31) == 0) sred[threadIdx.x >> 5] = s;
    __syncthreads();
    if (threadIdx.x == 0) {
        float t = 0.f;
#pragma unroll
        for (int i = 0; i < 8; ++i) t += sred[i];
        sq[row] = t;
        if (mask && mask[row]) atomicAdd(totalCnt, 1);
    }
}

// ---- main: fused f16 WMMA distance + forward/reverse argmin -------------
// Block = 128 tokens x all 4096 codes. Each wave owns a resident 16-token
// A-tile; all 8 waves share LDS-staged (double-buffered) B tiles.
__global__ __launch_bounds__(256)
void vq_dist_argmin(const _Float16* __restrict__ fh,
                    const _Float16* __restrict__ ch,
                    const float* __restrict__ f2,
                    const float* __restrict__ c2,
                    const int* __restrict__ mask,
                    unsigned long long* __restrict__ revKey,
                    int* __restrict__ assignWs,
                    float* __restrict__ assignOut)
{
    __shared__ _Float16 stage[2][16 * DF];        // 2 x 8KB B tiles
    __shared__ unsigned long long ldsRev[KC];     // 32KB per-code running min

    const int tid    = threadIdx.x;
    const int lane   = tid & 31;
    const int wave   = tid >> 5;
    const int laneLo = lane & 15;
    const int hi     = lane >> 4;
    const int tokW   = blockIdx.x * MTOK + wave * 16;   // this wave's tokens

    for (int i = tid; i < KC; i += 256) ldsRev[i] = ~0ull;

    // A fragments (16x32 f16 tiles over D=256), resident for the whole loop.
    // lane L: row M=L&15, K = hi*16..hi*16+15 contiguous.
    v16h a[8];
    const _Float16* ap = fh + (size_t)(tokW + laneLo) * DF + hi * 16;
#pragma unroll
    for (int f = 0; f < 8; ++f) {
        v8h lo = *(const v8h*)(ap + f * 32);
        v8h hv = *(const v8h*)(ap + f * 32 + 8);
        a[f] = __builtin_shufflevector(lo, hv, 0,1,2,3,4,5,6,7,8,9,10,11,12,13,14,15);
    }
    float rowBase[8];
#pragma unroll
    for (int r = 0; r < 8; ++r) {
        int tr = tokW + hi * 8 + r;
        rowBase[r] = mask[tr] ? f2[tr] : __builtin_inff();
    }
    float bestv[8]; int besti[8];
#pragma unroll
    for (int r = 0; r < 8; ++r) { bestv[r] = __builtin_inff(); besti[r] = 0; }

    stage_tile(&stage[0][0], ch, tid);            // prefetch tile 0

    for (int t = 0; t < NT; ++t) {
        wait_stage();          // my async ops for tile t landed
        __syncthreads();       // everyone's landed; everyone done with t-1
        if (t + 1 < NT)
            stage_tile(&stage[(t + 1) & 1][0], ch + (size_t)(t + 1) * 16 * DF, tid);

        // B from LDS: lane L holds column N=L&15, K = hi*16..+15 contiguous.
        const _Float16* bp = &stage[t & 1][laneLo * DF + hi * 16];
        v8f c0 = {}, c1 = {};   // two chains -> shorter WMMA dependency chain
#pragma unroll
        for (int f = 0; f < 4; ++f) {
            v16h b0 = *(const v16h*)(bp + (2 * f) * 32);
            v16h b1 = *(const v16h*)(bp + (2 * f + 1) * 32);
            c0 = __builtin_amdgcn_wmma_f32_16x16x32_f16(false, a[2 * f],     false, b0,
                                                        (short)0, c0, false, false);
            c1 = __builtin_amdgcn_wmma_f32_16x16x32_f16(false, a[2 * f + 1], false, b1,
                                                        (short)0, c1, false, false);
        }

        const int   n   = t * 16 + laneLo;        // this lane's code column
        const float c2n = c2[n];
        float colMin = __builtin_inff(); int colTok = 0x7FFFFFFF;
        // branchless running mins; within-lane ties keep first index because
        // both code order (across tiles) and row order (r) are ascending
#pragma unroll
        for (int r = 0; r < 8; ++r) {
            float dot  = c0[r] + c1[r];
            float dist = rowBase[r] - 2.0f * dot + c2n;  // ||f||^2 - 2 f.c + ||c||^2
            bool fsel = dist < bestv[r];
            bestv[r] = fsel ? dist : bestv[r];
            besti[r] = fsel ? n    : besti[r];
            int  tokr = tokW + hi * 8 + r;
            bool csel = dist < colMin;
            colMin = csel ? dist : colMin;
            colTok = csel ? tokr : colTok;
        }
        // combine the two half-wave row groups; only hi==0 pushes, and its
        // local rows have strictly lower tokens -> '<' preserves tie rule
        float ov = __shfl_xor(colMin, 16, 32);
        int   oi = __shfl_xor(colTok, 16, 32);
        bool  hsel = ov < colMin;
        colMin = hsel ? ov : colMin;
        colTok = hsel ? oi : colTok;
        if (hi == 0)
            atomicMin(&ldsRev[n], packKey(colMin, (unsigned)colTok));
    }

    // forward argmin: wave-private (this wave saw ALL codes for its tokens)
#pragma unroll
    for (int r = 0; r < 8; ++r) {
        float v = bestv[r]; int i = besti[r];
#pragma unroll
        for (int off = 1; off < 16; off <<= 1) {
            float ov2 = __shfl_xor(v, off, 32);
            int   oi2 = __shfl_xor(i, off, 32);
            unsigned better = (unsigned)(ov2 < v) | ((unsigned)(ov2 == v) & (unsigned)(oi2 < i));
            v = better ? ov2 : v;
            i = better ? oi2 : i;
        }
        if (laneLo == 0) {
            int tokn = tokW + hi * 8 + r;
            assignWs[tokn]  = i;
            assignOut[tokn] = (float)i;
        }
    }

    // flush per-block reverse-argmin table: one global atomic per code
    __syncthreads();
    for (int i = tid; i < KC; i += 256)
        atomicMin(&revKey[i], ldsRev[i]);
}

// ---- scatter: out_features, feat_sum, assignment_count, commitment ------
__global__ __launch_bounds__(256)
void vq_scatter(const float* __restrict__ features, const float* __restrict__ codebook,
                const int* __restrict__ assign, const int* __restrict__ mask,
                float* __restrict__ outFeat, float* __restrict__ featSum,
                float* __restrict__ acount, float* __restrict__ commitAcc)
{
    __shared__ float sred[8];
    const int n = blockIdx.x, d = threadIdx.x;
    const int j = assign[n];
    float f  = features[(size_t)n * DF + d];
    float cb = codebook[(size_t)j * DF + d];
    outFeat[(size_t)n * DF + d] = (cb + f) - f;       // STE, same fp order as source
    atomicAdd(&featSum[(size_t)j * DF + d], f);       // unmasked, as in source
    float diff = cb - f;
    float s = diff * diff;
#pragma unroll
    for (int off = 16; off > 0; off >>= 1) s += __shfl_xor(s, off, 32);
    if ((threadIdx.x & 31) == 0) sred[threadIdx.x >> 5] = s;
    __syncthreads();
    if (threadIdx.x == 0) {
        float t = 0.f;
#pragma unroll
        for (int i = 0; i < 8; ++i) t += sred[i];
        if (mask[n]) {
            atomicAdd(commitAcc, t * (1.0f / DF));
            atomicAdd(&acount[j], 1.0f);
        }
    }
}

// ---- codebook / count / avg update --------------------------------------
__global__ __launch_bounds__(256)
void vq_update(const float* __restrict__ features, const float* __restrict__ codebook,
               const float* __restrict__ countIn, const float* __restrict__ avgIn,
               const float* __restrict__ acount, const float* __restrict__ featSum,
               const unsigned long long* __restrict__ revKey,
               const int* __restrict__ totalCnt,
               float* __restrict__ outCount, float* __restrict__ outAvg,
               float* __restrict__ outCodebook, int* __restrict__ assignedCnt)
{
    const int k = blockIdx.x, d = threadIdx.x;
    const float acnt   = acount[k];
    const float totalF = fmaxf((float)(*totalCnt), 1.0f);
    const float cnew   = (1.0f - GAMMA_F) * acnt + GAMMA_F * countIn[k];
    const float anew   = (1.0f - GAMMA_F) * acnt / totalF + GAMMA_F * avgIn[k];
    const float alpha  = expf(-anew * (float)KC * 10.0f / (1.0f - GAMMA_F) - 0.001f);
    unsigned tok = (unsigned)(revKey[k] & 0xFFFFFFFFull);
    if (tok >= NTOK) tok = 0;    // safety clamp
    const float cb = codebook[(size_t)k * DF + d];
    const float fs = featSum[(size_t)k * DF + d];
    const float assigned = (GAMMA_F * cb + (1.0f - GAMMA_F) * fs) / fmaxf(cnew, 1.0f);
    const float frev     = features[(size_t)tok * DF + d];
    const float unass    = (1.0f - alpha) * cb + alpha * frev;
    // Source selects assigned_update where count<1 (replicated verbatim)
    const float newc = (acnt < 1.0f) ? assigned : unass;
    outCodebook[(size_t)k * DF + d] = 2.0f * cb - newc;   // prev + (prev - new)
    if (d == 0) {
        outCount[k] = cnew;
        outAvg[k]   = anew;
        if (acnt > 0.0f) atomicAdd(assignedCnt, 1);
    }
}

__global__ void vq_finalize(const float* __restrict__ commitAcc,
                            const int* __restrict__ totalCnt,
                            const int* __restrict__ assignedCnt,
                            float* __restrict__ outLoss, float* __restrict__ outPct)
{
    if (threadIdx.x == 0) {
        float totalF = fmaxf((float)(*totalCnt), 1.0f);
        *outLoss = *commitAcc / totalF;
        *outPct  = (float)(*assignedCnt) / (float)KC;
    }
}

extern "C" void kernel_launch(void* const* d_in, const int* in_sizes, int n_in,
                              void* d_out, int out_size, void* d_ws, size_t ws_size,
                              hipStream_t stream)
{
    const float* features = (const float*)d_in[0];
    const int*   mask     = (const int*)  d_in[1];
    const float* codebook = (const float*)d_in[2];
    const float* countIn  = (const float*)d_in[3];
    const float* avgIn    = (const float*)d_in[4];

    char* ws = (char*)d_ws;
    size_t off = 0;
    _Float16* fh = (_Float16*)(ws + off); off += (size_t)NTOK * DF * 2;
    _Float16* ch = (_Float16*)(ws + off); off += (size_t)KC   * DF * 2;
    float* featSum = (float*)(ws + off);  off += (size_t)KC   * DF * 4;
    float* f2      = (float*)(ws + off);  off += (size_t)NTOK * 4;
    float* c2      = (float*)(ws + off);  off += (size_t)KC   * 4;
    unsigned long long* revKey = (unsigned long long*)(ws + off); off += (size_t)KC * 8;
    int*   assignWs = (int*)(ws + off);   off += (size_t)NTOK * 4;
    float* acount   = (float*)(ws + off); off += (size_t)KC * 4;
    int*   totalCnt    = (int*)  (ws + off);
    float* commitAcc   = (float*)(ws + off + 4);
    int*   assignedCnt = (int*)  (ws + off + 8);

    float* outFeat     = (float*)d_out;
    float* outAssign   = outFeat   + (size_t)NTOK * DF;
    float* outLoss     = outAssign + NTOK;
    float* outPct      = outLoss   + 1;
    float* outCount    = outPct    + 1;
    float* outAvg      = outCount  + KC;
    float* outCodebook = outAvg    + KC;

    // capture-safe re-init of all accumulators every launch
    (void)hipMemsetAsync(revKey,  0xFF, (size_t)KC * 8, stream);
    (void)hipMemsetAsync(featSum, 0,    (size_t)KC * DF * 4, stream);
    (void)hipMemsetAsync(acount,  0,    (size_t)KC * 4, stream);
    (void)hipMemsetAsync(totalCnt,0,    16, stream);

    vq_prep_rows<<<NTOK, 256, 0, stream>>>(features, fh, f2, mask, totalCnt);
    vq_prep_rows<<<KC,   256, 0, stream>>>(codebook, ch, c2, nullptr, nullptr);
    vq_dist_argmin<<<NTOK / MTOK, 256, 0, stream>>>(fh, ch, f2, c2, mask,
                                                    revKey, assignWs, outAssign);
    vq_scatter<<<NTOK, 256, 0, stream>>>(features, codebook, assignWs, mask,
                                         outFeat, featSum, acount, commitAcc);
    vq_update<<<KC, 256, 0, stream>>>(features, codebook, countIn, avgIn,
                                      acount, featSum, revKey, totalCnt,
                                      outCount, outAvg, outCodebook, assignedCnt);
    vq_finalize<<<1, 32, 0, stream>>>(commitAcc, totalCnt, assignedCnt, outLoss, outPct);
}
